// Ibex_UCI_MLP_15049565405381
// MI455X (gfx1250) — matrix operand order repaired and checked by reading, not compiled, not run on hardware
//
#include <hip/hip_runtime.h>

// ---------------------------------------------------------------------------
// Fused 4-layer requantized MLP for MI455X (gfx1250, wave32, WMMA).
//   h = requant(X @ W.T + b) x4, requant -> integers in [0,255] (f16-exact).
// One 128-thread block (4 waves) owns 64 batch rows; each wave owns 16 rows
// end-to-end. Activations stay in LDS, weights pre-packed into WMMA B-frag
// layout (f16 hi/lo split for f32-class accuracy), f32 accumulation via
// v_wmma_f32_16x16x32_f16. Biases pre-padded so the WMMA loop is fully
// divergence-free (EXEC stays all-1s, as WMMA requires). Requant uses
// contraction-proof mul/add to match the reference's two-rounding f32 math.
// ---------------------------------------------------------------------------

typedef __attribute__((ext_vector_type(16))) _Float16 v16h;
typedef __attribute__((ext_vector_type(8)))  _Float16 v8h;
typedef __attribute__((ext_vector_type(8)))  float    v8f;

constexpr int BATCH  = 262144;
constexpr int APITCH = 328;   // f16 pitch of main LDS activation buffer (>=320, 16B-multiple rows)
constexpr int XPITCH = 104;   // f16 pitch of layer-0 lo-residual buffer (>=96)

// Opaque single-instruction f32 ops: the reference (jnp, f32) computes x*m and
// +2^14 as two separately-rounded ops; these stop clang from fusing them into
// v_fma_f32 (single rounding), which could flip trunc/floor boundaries.
__device__ __forceinline__ float fmul_sep(float a, float b) {
  float r;
  asm("v_mul_f32_e32 %0, %1, %2" : "=v"(r) : "v"(a), "v"(b));
  return r;
}
__device__ __forceinline__ float fadd_sep(float a, float b) {
  float r;
  asm("v_add_f32_e32 %0, %1, %2" : "=v"(r) : "v"(a), "v"(b));
  return r;
}

// requant: t = trunc(x*m + 2^14); clip(floor(t * 2^-15), 0, 255)
__device__ __forceinline__ float requant(float x, float mul) {
  const float p = fmul_sep(x, mul);
  const float t = truncf(fadd_sep(p, 16384.0f));
  float q = floorf(t * 3.0517578125e-05f); // * 2^-15 (exact pow2, mul-only)
  return fminf(fmaxf(q, 0.0f), 255.0f);
}

// ---------------------------------------------------------------------------
// Weight packer: W[N][K] f32 -> per-fragment WMMA B layout, f16 hi + f16 lo.
// Fragment (kt,nt) covers K rows [kt*32, kt*32+32), N cols [nt*16, nt*16+16).
// B 32x16 f16 layout: lane half h = col n, elems 0-7 = K kt*32+8h..+7,
// elems 8-15 = K kt*32+16+8h..+7  => each lane owns 16 f16 (32 bytes).
// Pair stride = 1024 f16 (hi at +0, lo at +512).
// ---------------------------------------------------------------------------
__global__ void pack_w(const float* __restrict__ W, _Float16* __restrict__ out,
                       int K, int N, int KT, int NT) {
  const int idx = blockIdx.x * blockDim.x + threadIdx.x;
  const int total = KT * NT * 32;
  if (idx >= total) return;
  const int lane = idx & 31;
  const int f    = idx >> 5;          // fragment index = kt*NT + nt
  const int nt   = f % NT;
  const int kt   = f / NT;
  const int n     = nt * 16 + (lane & 15);
  const int kbase = kt * 32 + (lane >> 4) * 8;
  _Float16* hi = out + (size_t)f * 1024 + lane * 16;
  _Float16* lo = hi + 512;
  #pragma unroll
  for (int j = 0; j < 16; ++j) {
    const int k = kbase + j + (j >= 8 ? 8 : 0);
    const float w = (k < K && n < N) ? W[(size_t)n * K + k] : 0.0f;
    const _Float16 h = (_Float16)w;
    hi[j] = h;
    lo[j] = (_Float16)(w - (float)h);
  }
}

// Bias padder: b[N] f32 -> out[NPAD] f32 with zeros in the pad region, so the
// GEMM epilogue needs no bounds check (requant(0 + 0) == 0 keeps padding 0).
__global__ void pack_bias(const float* __restrict__ b, float* __restrict__ out,
                          int N, int NPAD) {
  const int i = blockIdx.x * blockDim.x + threadIdx.x;
  if (i < NPAD) out[i] = (i < N) ? b[i] : 0.0f;
}

// ---------------------------------------------------------------------------
// One layer for one wave. A frags read from LDS (per ISA 7.12.2 16-bit A
// layout: two 16B chunks per lane), all KT frags hoisted to VGPRs, then the
// nt loop runs KT (x2 or x3 for split) WMMAs and a requant epilogue that
// writes back to LDS (or global for the last layer). No divergence in the
// nt loop except the final-layer global store guard.
// ---------------------------------------------------------------------------
template <int KT, int NT, int NCOLS, bool ASPLIT, bool LAST>
__device__ __forceinline__ void mlp_layer(
    _Float16* sact, const _Float16* sxlo,
    const _Float16* __restrict__ wpk, const float* __restrict__ bias_pad,
    float mul, int lane, int wrow0, long m0, float* __restrict__ gout) {
  const int r    = lane & 15;
  const int half = lane >> 4;

  v16h a[KT];
  v16h al[ASPLIT ? KT : 1];
  const _Float16* rp = sact + (size_t)(wrow0 + r) * APITCH;
  #pragma unroll
  for (int kt = 0; kt < KT; ++kt) {
    union { v16h v; v8h h[2]; } u;
    u.h[0] = *(const v8h*)(rp + kt * 32 + half * 8);        // K kt*32 + half*8 ..+7
    u.h[1] = *(const v8h*)(rp + kt * 32 + 16 + half * 8);   // K kt*32+16+half*8 ..+7
    a[kt] = u.v;
  }
  if constexpr (ASPLIT) {
    const _Float16* rq = sxlo + (size_t)(wrow0 + r) * XPITCH;
    #pragma unroll
    for (int kt = 0; kt < KT; ++kt) {
      union { v16h v; v8h h[2]; } u;
      u.h[0] = *(const v8h*)(rq + kt * 32 + half * 8);
      u.h[1] = *(const v8h*)(rq + kt * 32 + 16 + half * 8);
      al[kt] = u.v;
    }
  }

  for (int nt = 0; nt < NT; ++nt) {
    v8f acc = {};
    #pragma unroll
    for (int kt = 0; kt < KT; ++kt) {
      const _Float16* fp = wpk + (((size_t)(kt * NT + nt)) << 10) + lane * 16;
      const v16h bhi = *(const v16h*)fp;
      const v16h blo = *(const v16h*)(fp + 512);
      // acc += A_hi*B_lo (+ A_lo*B_hi) + A_hi*B_hi
      acc = __builtin_amdgcn_wmma_f32_16x16x32_f16(false, a[kt], false, blo,
                                                   (short)0, acc, false, false);
      if constexpr (ASPLIT)
        acc = __builtin_amdgcn_wmma_f32_16x16x32_f16(false, al[kt], false, bhi,
                                                     (short)0, acc, false, false);
      acc = __builtin_amdgcn_wmma_f32_16x16x32_f16(false, a[kt], false, bhi,
                                                   (short)0, acc, false, false);
    }
    // C/D layout: VGPR i -> row (i + half*8), lane -> col (nt*16 + r)
    const int col = nt * 16 + r;
    const float bv = bias_pad[col];   // padded: no bounds check, no EXEC churn
    #pragma unroll
    for (int i = 0; i < 8; ++i) {
      const float q = requant(acc[i] + bv, mul);
      const int row = i + half * 8;
      if constexpr (LAST) {
        if (col < NCOLS) gout[(m0 + row) * NCOLS + col] = q;
      } else {
        sact[(size_t)(wrow0 + row) * APITCH + col] = (_Float16)q;
      }
    }
  }
}

// ---------------------------------------------------------------------------
// Fused MLP kernel. grid = BATCH/64 blocks of 128 threads (4 waves).
// Waves never share LDS rows -> no barriers; per-wave LDS ordering covers the
// store->load dependency between layers.
// ---------------------------------------------------------------------------
__global__ __launch_bounds__(128) void fused_mlp(
    const float* __restrict__ X,
    const _Float16* __restrict__ w0p, const float* __restrict__ b0p,
    const _Float16* __restrict__ w1p, const float* __restrict__ b1p,
    const _Float16* __restrict__ w2p, const float* __restrict__ b2p,
    const _Float16* __restrict__ w3p, const float* __restrict__ b3p,
    float* __restrict__ out) {
  __shared__ _Float16 sact[64 * APITCH];  // 41.0 KB: activations, all layers
  __shared__ _Float16 sxlo[64 * XPITCH];  // 13.0 KB: layer-0 f16 residual of X

  const int tid   = threadIdx.x;
  const int lane  = tid & 31;
  const int wave  = tid >> 5;
  const int wrow0 = wave * 16;
  const long m0   = (long)blockIdx.x * 64 + wrow0;

  // Stage this wave's 16 rows of X as f16 hi/lo, zero-padded K 76 -> 96.
  // 16*96 = 1536 elements, 48 uniform iterations per lane (EXEC stays full).
  for (int idx = lane; idx < 16 * 96; idx += 32) {
    const int row = idx / 96;
    const int c   = idx - row * 96;
    const float x = (c < 76) ? X[(m0 + row) * 76 + c] : 0.0f;
    const _Float16 h = (_Float16)x;
    sact[(size_t)(wrow0 + row) * APITCH + c] = h;
    sxlo[(size_t)(wrow0 + row) * XPITCH + c] = (_Float16)(x - (float)h);
  }

  // KPADs: 96, 320, 224, 128  ->  KT: 3, 10, 7, 4
  // N padded to next layer's KPAD -> NT: 20, 14, 8, 1 (pad cols requant to 0)
  mlp_layer<3, 20, 300, true,  false>(sact, sxlo,    w0p, b0p, 1500.0f, lane, wrow0, m0, nullptr);
  mlp_layer<10, 14, 200, false, false>(sact, nullptr, w1p, b1p, 1400.0f, lane, wrow0, m0, nullptr);
  mlp_layer<7,  8,  100, false, false>(sact, nullptr, w2p, b2p, 1300.0f, lane, wrow0, m0, nullptr);
  mlp_layer<4,  1,  10,  false, true >(sact, nullptr, w3p, b3p, 1200.0f, lane, wrow0, m0, out);
}

// ---------------------------------------------------------------------------
// Host launcher. Workspace: packed hi/lo weights (~532 KB, L2-resident) plus
// padded biases (2.7 KB).
// ---------------------------------------------------------------------------
extern "C" void kernel_launch(void* const* d_in, const int* in_sizes, int n_in,
                              void* d_out, int out_size, void* d_ws, size_t ws_size,
                              hipStream_t stream) {
  (void)in_sizes; (void)n_in; (void)out_size; (void)ws_size;
  const float* X  = (const float*)d_in[0];
  const float* W0 = (const float*)d_in[1];  const float* b0 = (const float*)d_in[2];
  const float* W1 = (const float*)d_in[3];  const float* b1 = (const float*)d_in[4];
  const float* W2 = (const float*)d_in[5];  const float* b2 = (const float*)d_in[6];
  const float* W3 = (const float*)d_in[7];  const float* b3 = (const float*)d_in[8];

  _Float16* ws  = (_Float16*)d_ws;
  _Float16* w0p = ws;                        // 3*20 frag-pairs * 1024 f16
  _Float16* w1p = w0p + (size_t)3  * 20 * 1024;
  _Float16* w2p = w1p + (size_t)10 * 14 * 1024;
  _Float16* w3p = w2p + (size_t)7  * 8  * 1024;
  float*    b0p = (float*)(w3p + (size_t)4 * 1 * 1024);  // 4B-aligned (even f16 count)
  float*    b1p = b0p + 320;
  float*    b2p = b1p + 224;
  float*    b3p = b2p + 128;                 // +16 -> end of workspace usage

  pack_w<<<(3 * 20 * 32 + 127) / 128, 128, 0, stream>>>(W0, w0p, 76, 300, 3, 20);
  pack_w<<<(10 * 14 * 32 + 127) / 128, 128, 0, stream>>>(W1, w1p, 300, 200, 10, 14);
  pack_w<<<(7 * 8 * 32 + 127) / 128, 128, 0, stream>>>(W2, w2p, 200, 100, 7, 8);
  pack_w<<<1, 128, 0, stream>>>(W3, w3p, 100, 10, 4, 1);
  pack_bias<<<3, 128, 0, stream>>>(b0, b0p, 300, 320);
  pack_bias<<<2, 128, 0, stream>>>(b1, b1p, 200, 224);
  pack_bias<<<1, 128, 0, stream>>>(b2, b2p, 100, 128);
  pack_bias<<<1, 128, 0, stream>>>(b3, b3p, 10, 16);

  fused_mlp<<<BATCH / 64, 128, 0, stream>>>(X, w0p, b0p, w1p, b1p, w2p, b2p,
                                            w3p, b3p, (float*)d_out);
}
